// MemoryLayerCell_26860725469502
// MI455X (gfx1250) — matrix operand order repaired and checked by reading, not compiled, not run on hardware
//
#include <hip/hip_runtime.h>
#include <hip/hip_bf16.h>
#include <cstdint>

typedef __attribute__((ext_vector_type(16))) __bf16 v16bf;
typedef __attribute__((ext_vector_type(8)))  __bf16 v8bf;
typedef __attribute__((ext_vector_type(8)))  float  v8f;

#define TILE_M 128
#define TILE_N 128
#define TILE_K 32
#define LDSL   40   // padded bf16 stride per row (80 B) to spread LDS banks

// ---------------------------------------------------------------------------
// CDNA5 async global->LDS copy (ASYNCcnt path). LDS operand is the wave-
// relative LDS byte offset = low 32 bits of the generic pointer to __shared__.
// ---------------------------------------------------------------------------
__device__ __forceinline__ uint32_t lds_off(const void* p)
{
    return (uint32_t)(uintptr_t)p;
}

__device__ __forceinline__ void async_copy_b128(uint32_t lds, const void* gptr)
{
#if defined(__HIP_DEVICE_COMPILE__)
    asm volatile("global_load_async_to_lds_b128 %0, %1, off"
                 :: "v"(lds), "v"(gptr) : "memory");
#endif
}

__device__ __forceinline__ void wait_async0()
{
#if defined(__HIP_DEVICE_COMPILE__)
#if __has_builtin(__builtin_amdgcn_s_wait_asynccnt)
    __builtin_amdgcn_s_wait_asynccnt(0);
#else
    asm volatile("s_wait_asynccnt 0x0" ::: "memory");
#endif
#endif
}

// ---------------------------------------------------------------------------
// Pack [inputs | states] rows into bf16 A matrix (B x 1280)
// ---------------------------------------------------------------------------
__global__ void pack_a(const float* __restrict__ inputs,
                       const float* __restrict__ states,
                       __bf16* __restrict__ A, int total)
{
    for (int idx = blockIdx.x * blockDim.x + threadIdx.x; idx < total;
         idx += gridDim.x * blockDim.x) {
        int row = idx / 1280;
        int col = idx - row * 1280;
        float v = (col < 256) ? inputs[(size_t)row * 256 + col]
                              : states[(size_t)row * 1024 + (col - 256)];
        A[idx] = (__bf16)v;
    }
}

// ---------------------------------------------------------------------------
// LDS-tiled transpose+convert:  out[n*K + k] = bf16(in[k*N + n])  (W is K x N)
// Grid: (K/32, N/32), 256 threads. K,N are multiples of 32 for all weights.
// ---------------------------------------------------------------------------
__global__ __launch_bounds__(256)
void packT(const float* __restrict__ in, __bf16* __restrict__ out, int K, int N)
{
    __shared__ float tile[32][33];
    const int kb = blockIdx.x * 32, nb = blockIdx.y * 32;
    const int tx = threadIdx.x & 31, ty = threadIdx.x >> 5;   // ty = 0..7
#pragma unroll
    for (int yy = 0; yy < 32; yy += 8)
        tile[ty + yy][tx] = in[(size_t)(kb + ty + yy) * N + nb + tx];
    __syncthreads();
#pragma unroll
    for (int yy = 0; yy < 32; yy += 8)
        out[(size_t)(nb + ty + yy) * K + kb + tx] = (__bf16)tile[tx][ty + yy];
}

// ---------------------------------------------------------------------------
// Tiled bf16 WMMA GEMM:  C[M,N] = A[M,K] * Wt[N,K]^T + bias
// 256 threads = 8 waves; block tile 128x128, K-chunk 32, double-buffered LDS
// filled by global_load_async_to_lds_b128 (no VGPR staging, one barrier/iter).
// Wave grid: 4 along M (32 rows) x 2 along N (64 cols); 8 WMMA accums/wave.
// ---------------------------------------------------------------------------
template <typename OutT>
__global__ __launch_bounds__(256)
void gemm_bf16_wmma(const __bf16* __restrict__ A, const __bf16* __restrict__ Wt,
                    const float* __restrict__ bias, OutT* __restrict__ C,
                    int M, int N, int K)
{
    __shared__ __bf16 Asl[2][TILE_M][LDSL];
    __shared__ __bf16 Bsl[2][TILE_N][LDSL];

    const int tid  = threadIdx.x;
    const int lane = tid & 31;
    const int wave = tid >> 5;
    const int wm   = wave & 3;        // M offset = wm*32
    const int wn   = wave >> 2;       // N offset = wn*64
    const int m0   = blockIdx.y * TILE_M;
    const int n0   = blockIdx.x * TILE_N;

    const v8f vzero = {0.f, 0.f, 0.f, 0.f, 0.f, 0.f, 0.f, 0.f};
    v8f acc[2][4];
#pragma unroll
    for (int mt = 0; mt < 2; ++mt)
#pragma unroll
        for (int nt = 0; nt < 4; ++nt) acc[mt][nt] = vzero;

    // staging: 512 chunks of 8 bf16 (16 B); thread t -> rows t>>2 and t>>2+64,
    // column (t&3)*8
    const int ar0 = tid >> 2;
    const int ar1 = ar0 + 64;
    const int ac  = (tid & 3) * 8;

    const __bf16* ga0 = A  + (size_t)(m0 + ar0) * K + ac;
    const __bf16* ga1 = A  + (size_t)(m0 + ar1) * K + ac;
    const __bf16* gw0 = Wt + (size_t)(n0 + ar0) * K + ac;
    const __bf16* gw1 = Wt + (size_t)(n0 + ar1) * K + ac;

    uint32_t la0[2], la1[2], lb0[2], lb1[2];
#pragma unroll
    for (int b = 0; b < 2; ++b) {
        la0[b] = lds_off(&Asl[b][ar0][ac]);
        la1[b] = lds_off(&Asl[b][ar1][ac]);
        lb0[b] = lds_off(&Bsl[b][ar0][ac]);
        lb1[b] = lds_off(&Bsl[b][ar1][ac]);
    }

    // prologue: tile 0 -> buffer 0
    async_copy_b128(la0[0], ga0);
    async_copy_b128(la1[0], ga1);
    async_copy_b128(lb0[0], gw0);
    async_copy_b128(lb1[0], gw1);

    const int h   = lane >> 4;   // K-half select per ISA 16-bit fragment layout
    const int m16 = lane & 15;
    const int nk  = K / TILE_K;

    for (int kt = 0; kt < nk; ++kt) {
        const int cur = kt & 1;
        wait_async0();           // my copies into buf[cur] have landed
        __syncthreads();         // everyone's copies landed; buf[cur^1] free

        if (kt + 1 < nk) {       // issue next tile into the other buffer
            const size_t ko  = (size_t)(kt + 1) * TILE_K;
            const int    nxt = cur ^ 1;
            async_copy_b128(la0[nxt], ga0 + ko);
            async_copy_b128(la1[nxt], ga1 + ko);
            async_copy_b128(lb0[nxt], gw0 + ko);
            async_copy_b128(lb1[nxt], gw1 + ko);
        }

        // A fragment (16x32 bf16): lanes 0-15 -> K 0..7 / 16..23,
        //                          lanes 16-31 -> K 8..15 / 24..31
        v16bf afrag[2];
#pragma unroll
        for (int mt = 0; mt < 2; ++mt) {
            const __bf16* ap = &Asl[cur][wm * 32 + mt * 16 + m16][0];
            v8bf lo = *(const v8bf*)(ap + h * 8);
            v8bf hi = *(const v8bf*)(ap + 16 + h * 8);
            afrag[mt] = __builtin_shufflevector(lo, hi, 0, 1, 2, 3, 4, 5, 6, 7,
                                                8, 9, 10, 11, 12, 13, 14, 15);
        }
        // B fragment (32x16 bf16): lane = N col, lanes 0-15 K 0..7 / 16..23, 16-31 K 8..15 / 24..31
        v16bf bfrag[4];
#pragma unroll
        for (int nt = 0; nt < 4; ++nt) {
            const __bf16* bp = &Bsl[cur][wn * 64 + nt * 16 + m16][h * 8];
            v8bf lo = *(const v8bf*)(bp);
            v8bf hi = *(const v8bf*)(bp + 16);
            bfrag[nt] = __builtin_shufflevector(lo, hi, 0, 1, 2, 3, 4, 5, 6, 7,
                                                8, 9, 10, 11, 12, 13, 14, 15);
        }
#pragma unroll
        for (int mt = 0; mt < 2; ++mt)
#pragma unroll
            for (int nt = 0; nt < 4; ++nt)
                acc[mt][nt] = __builtin_amdgcn_wmma_f32_16x16x32_bf16(
                    false, afrag[mt], false, bfrag[nt],
                    (short)0, acc[mt][nt], false, false);
    }

    // Epilogue: C/D layout — VGPR r, lanes<16 row r, lanes>=16 row r+8
    const int cn = lane & 15;
#pragma unroll
    for (int nt = 0; nt < 4; ++nt) {
        const int gn = n0 + wn * 64 + nt * 16 + cn;
        const float bv = bias[gn];
#pragma unroll
        for (int mt = 0; mt < 2; ++mt) {
#pragma unroll
            for (int r = 0; r < 8; ++r) {
                const int gm = m0 + wm * 32 + mt * 16 + 8 * h + r;
                C[(size_t)gm * N + gn] = (OutT)(acc[mt][nt][r] + bv);
            }
        }
    }
}

// ---------------------------------------------------------------------------
// Per-row conductance update + LayerNorm. Parameters staged once per block in
// LDS; each block (256 thr) processes 8 batch rows to amortize broadcast reads.
// presyn[j] = (s_j, s_{j^1}) for both parities; mci[j] = +/- pre[j>>1].
// ---------------------------------------------------------------------------
#define CELL_ROWS 8
__global__ __launch_bounds__(256)
void cell_kernel(const float* __restrict__ states, const float* __restrict__ pre,
                 const float* __restrict__ exc, const float* __restrict__ inh,
                 const float* __restrict__ cap, const float* __restrict__ maxg,
                 const float* __restrict__ meang, const float* __restrict__ stdg,
                 const float* __restrict__ gamma, const float* __restrict__ beta,
                 float* __restrict__ next_states, __bf16* __restrict__ cell_out)
{
    __shared__ float p1[5][1024];    // exc, inh, cap, gamma, beta
    __shared__ float p2[3][2048];    // maxg, meang, stdg  (S,2 interleaved)
    __shared__ float wsum[8], wssq[8];

    const int tid = threadIdx.x;
    for (int i = tid; i < 1024; i += 256) {
        p1[0][i] = exc[i];  p1[1][i] = inh[i];  p1[2][i] = cap[i];
        p1[3][i] = gamma[i]; p1[4][i] = beta[i];
    }
    for (int i = tid; i < 2048; i += 256) {
        p2[0][i] = maxg[i]; p2[1][i] = meang[i]; p2[2][i] = stdg[i];
    }
    __syncthreads();

    const int lane = tid & 31, wave = tid >> 5;
    for (int r = 0; r < CELL_ROWS; ++r) {
        const int b = blockIdx.x * CELL_ROWS + r;
        const float* srow = states + (size_t)b * 1024;
        const float* prow = pre    + (size_t)b * 512;

        float xs[4];
        float sum = 0.f, ssq = 0.f;
#pragma unroll
        for (int u = 0; u < 4; ++u) {
            const int j = tid * 4 + u;
            const float s  = srow[j];
            const float sp = srow[j ^ 1];
            const float mci = (j & 1) ? -prow[j >> 1] : prow[j >> 1];
            const float g0 = p2[0][2 * j]     / (1.f + __expf(-p2[2][2 * j]     * (s  - p2[1][2 * j])));
            const float g1 = p2[0][2 * j + 1] / (1.f + __expf(-p2[2][2 * j + 1] * (sp - p2[1][2 * j + 1])));
            const float x = s + (mci + g0 * (p1[0][j] - s) + g1 * (p1[1][j] - s)) / p1[2][j];
            xs[u] = x; sum += x; ssq += x * x;
        }
        // wave32 reduce, then cross-wave via LDS
#pragma unroll
        for (int off = 16; off; off >>= 1) {
            sum += __shfl_xor(sum, off, 32);
            ssq += __shfl_xor(ssq, off, 32);
        }
        if (lane == 0) { wsum[wave] = sum; wssq[wave] = ssq; }
        __syncthreads();
        float tsum = 0.f, tssq = 0.f;
#pragma unroll
        for (int w = 0; w < 8; ++w) { tsum += wsum[w]; tssq += wssq[w]; }
        const float mu  = tsum * (1.f / 1024.f);
        const float var = tssq * (1.f / 1024.f) - mu * mu;
        const float rs  = rsqrtf(var + 1e-6f);

        float*  nrow = next_states + (size_t)b * 1024;
        __bf16* crow = cell_out    + (size_t)b * 512;
#pragma unroll
        for (int u = 0; u < 4; ++u) {
            const int j = tid * 4 + u;
            const float ns = (xs[u] - mu) * rs * p1[3][j] + p1[4][j];
            nrow[j] = ns;
            if (!(u & 1)) crow[j >> 1] = (__bf16)ns;   // even state indices
        }
        __syncthreads();   // protect wsum/wssq reuse next row
    }
}

// ---------------------------------------------------------------------------
extern "C" void kernel_launch(void* const* d_in, const int* in_sizes, int n_in,
                              void* d_out, int out_size, void* d_ws, size_t ws_size,
                              hipStream_t stream)
{
    const float* inputs = (const float*)d_in[0];
    const float* states = (const float*)d_in[1];
    const float* W1  = (const float*)d_in[2];
    const float* b1  = (const float*)d_in[3];
    const float* W2  = (const float*)d_in[4];
    const float* b2  = (const float*)d_in[5];
    const float* W3  = (const float*)d_in[6];
    const float* b3  = (const float*)d_in[7];
    const float* W4  = (const float*)d_in[8];
    const float* b4  = (const float*)d_in[9];
    const float* exc = (const float*)d_in[10];
    const float* inh = (const float*)d_in[11];
    const float* cap = (const float*)d_in[12];
    const float* mxg = (const float*)d_in[13];
    const float* mng = (const float*)d_in[14];
    const float* sdg = (const float*)d_in[15];
    const float* gam = (const float*)d_in[16];
    const float* bet = (const float*)d_in[17];

    constexpr int Bb = 16384, Dd = 256, Ss = 1024, Oo = 256;

    char* ws = (char*)d_ws;
    size_t off = 0;
    // region 0: A (bf16, B x 1280) ... later reused as pre (f32, B x 512)
    __bf16* Abf = (__bf16*)(ws + off);
    float*  pre = (float*)(ws + off);
    off += (size_t)Bb * (Dd + Ss) * 2;                 // 41,943,040 B
    __bf16* W1t = (__bf16*)(ws + off); off += (size_t)1024 * 1280 * 2;
    __bf16* W2t = (__bf16*)(ws + off); off += (size_t)512 * 1024 * 2;
    __bf16* W3t = (__bf16*)(ws + off); off += (size_t)512 * 512 * 2;
    __bf16* W4t = (__bf16*)(ws + off); off += (size_t)256 * 512 * 2;
    // region 2: h (bf16, B x 1024) ... later reused as cell (B x 512) + t3 (B x 512)
    __bf16* hbf  = (__bf16*)(ws + off);
    __bf16* cell = hbf;
    __bf16* t3   = (__bf16*)(ws + off + (size_t)Bb * 512 * 2);

    float* out  = (float*)d_out;                 // (B, 256)
    float* next = out + (size_t)Bb * Oo;         // (B, 1024)

    // 1) pack activations + weights to bf16 (weights transposed to N x K)
    {
        int total = Bb * (Dd + Ss);
        pack_a<<<dim3((total + 1023) / 1024), dim3(256), 0, stream>>>(inputs, states, Abf, total);
    }
    packT<<<dim3(1280 / 32, 1024 / 32), dim3(256), 0, stream>>>(W1, W1t, 1280, 1024);
    packT<<<dim3(1024 / 32,  512 / 32), dim3(256), 0, stream>>>(W2, W2t, 1024, 512);
    packT<<<dim3( 512 / 32,  512 / 32), dim3(256), 0, stream>>>(W3, W3t, 512, 512);
    packT<<<dim3( 512 / 32,  256 / 32), dim3(256), 0, stream>>>(W4, W4t, 512, 256);

    // 2) h = [x,s] @ W1 + b1   (bf16 out -> feeds GEMM2)
    gemm_bf16_wmma<__bf16><<<dim3(1024 / TILE_N, Bb / TILE_M), 256, 0, stream>>>(
        Abf, W1t, b1, hbf, Bb, 1024, 1280);
    // 3) pre = h @ W2 + b2     (f32 out -> feeds elementwise)
    gemm_bf16_wmma<float><<<dim3(512 / TILE_N, Bb / TILE_M), 256, 0, stream>>>(
        hbf, W2t, b2, pre, Bb, 512, 1024);
    // 4) conductance update + LayerNorm -> next_states (d_out), cell (bf16)
    cell_kernel<<<dim3(Bb / CELL_ROWS), dim3(256), 0, stream>>>(
        states, pre, exc, inh, cap, mxg, mng, sdg, gam, bet, next, cell);
    // 5) t3 = cell @ W3 + b3   (bf16 out)
    gemm_bf16_wmma<__bf16><<<dim3(512 / TILE_N, Bb / TILE_M), 256, 0, stream>>>(
        cell, W3t, b3, t3, Bb, 512, 512);
    // 6) out = t3 @ W4 + b4    (f32 out -> d_out)
    gemm_bf16_wmma<float><<<dim3(256 / TILE_N, Bb / TILE_M), 256, 0, stream>>>(
        t3, W4t, b4, out, Bb, 256, 512);
}
